// MultiHeadCrossAttention_71373766525200
// MI455X (gfx1250) — matrix operand order repaired and checked by reading, not compiled, not run on hardware
//
#include <hip/hip_runtime.h>

// ---- types ------------------------------------------------------------
typedef __attribute__((ext_vector_type(16))) __bf16        v16bf;
typedef __attribute__((ext_vector_type(8)))  float         v8f;
typedef __attribute__((ext_vector_type(4)))  unsigned int  v4u;
typedef __attribute__((ext_vector_type(4)))  float         v4f;

union Frag { v16bf v; v4u q[2]; };

__device__ __forceinline__ unsigned short f2bf(float f) {
  unsigned int u = __float_as_uint(f);
  u += 0x7fffu + ((u >> 16) & 1u);          // round-to-nearest-even
  return (unsigned short)(u >> 16);
}
__device__ __forceinline__ unsigned int f2bf2(float lo, float hi) {
  return ((unsigned int)f2bf(hi) << 16) | (unsigned int)f2bf(lo);
}

static constexpr int B_ = 4, T_ = 2048, C_ = 1024, H_ = 16, D_ = 64;
static constexpr float LOG_THETA = 9.210340371976184f;   // ln(10000)

// ---- projection GEMM: out = A(MxK f32) * W(KxN f32), bf16 epilogue ----
// Software-pipelined: global loads for tile kt+1 issue before the WMMAs of
// tile kt, so HBM latency overlaps matrix math.
// mode 0: q-proj, RoPE, write outK [B,H,T,D]
// mode 1: kv-proj, n<C -> RoPE k [B,H,T,D]; n>=C -> v transposed [B,H,D,T]
__global__ __launch_bounds__(256) void gemm_proj(
    const float* __restrict__ A, const float* __restrict__ W,
    unsigned short* __restrict__ outK, unsigned short* __restrict__ outV,
    int N, int mode)
{
  __shared__ unsigned short As[128][40];
  __shared__ unsigned short Bs[128][40];   // W tile stored [n][k]

  const int tid  = threadIdx.x;
  const int lane = tid & 31;
  const int wave = tid >> 5;
  const int wm   = wave >> 1;              // 0..3 -> 32-row slab
  const int wn   = wave & 1;               // 0..1 -> 64-col slab
  const int m0   = blockIdx.y * 128;
  const int n0   = blockIdx.x * 128;
  const int hw   = lane >> 4;              // half-wave
  const int l15  = lane & 15;

  v8f acc[2][4];
  #pragma unroll
  for (int a = 0; a < 2; ++a)
    #pragma unroll
    for (int b = 0; b < 4; ++b) acc[a][b] = {};

  v4f ra[4], rw[4];
  auto loadregs = [&](int kt) {
    #pragma unroll
    for (int i = 0; i < 4; ++i) {
      ra[i] = *(const v4f*)(A + (size_t)(m0 + (tid >> 3) + i * 32) * C_ +
                            kt * 32 + (tid & 7) * 4);
      rw[i] = *(const v4f*)(W + (size_t)(kt * 32 + (tid >> 5) + i * 8) * N +
                            n0 + (tid & 31) * 4);
    }
  };
  auto storelds = [&]() {
    #pragma unroll
    for (int i = 0; i < 4; ++i) {
      int row = (tid >> 3) + i * 32, col = (tid & 7) * 4;
      *(unsigned int*)&As[row][col]     = f2bf2(ra[i].x, ra[i].y);
      *(unsigned int*)&As[row][col + 2] = f2bf2(ra[i].z, ra[i].w);
      int kr = (tid >> 5) + i * 8, nc = (tid & 31) * 4;
      Bs[nc + 0][kr] = f2bf(rw[i].x); Bs[nc + 1][kr] = f2bf(rw[i].y);
      Bs[nc + 2][kr] = f2bf(rw[i].z); Bs[nc + 3][kr] = f2bf(rw[i].w);
    }
  };

  constexpr int KT = C_ / 32;
  loadregs(0);
  for (int kt = 0; kt < KT; ++kt) {
    storelds();
    __syncthreads();
    if (kt + 1 < KT) loadregs(kt + 1);     // overlap next tile with WMMAs

    Frag af[2], bfr[4];
    #pragma unroll
    for (int mi = 0; mi < 2; ++mi) {
      int row = wm * 32 + mi * 16 + l15;
      int kb  = hw * 8;
      af[mi].q[0] = *(const v4u*)&As[row][kb];
      af[mi].q[1] = *(const v4u*)&As[row][kb + 16];
    }
    #pragma unroll
    for (int ni = 0; ni < 4; ++ni) {
      int col = wn * 64 + ni * 16 + l15;
      int kb  = hw * 8;
      bfr[ni].q[0] = *(const v4u*)&Bs[col][kb];
      bfr[ni].q[1] = *(const v4u*)&Bs[col][kb + 16];
    }
    #pragma unroll
    for (int mi = 0; mi < 2; ++mi)
      #pragma unroll
      for (int ni = 0; ni < 4; ++ni)
        acc[mi][ni] = __builtin_amdgcn_wmma_f32_16x16x32_bf16(
            false, af[mi].v, false, bfr[ni].v, (short)0, acc[mi][ni], false, false);
    __syncthreads();
  }

  // epilogue: RoPE + scatter to head layout
  #pragma unroll
  for (int mi = 0; mi < 2; ++mi) {
    #pragma unroll
    for (int ni = 0; ni < 4; ++ni) {
      int ng = n0 + wn * 64 + ni * 16 + l15;
      #pragma unroll
      for (int r = 0; r < 8; ++r) {
        int mg = m0 + wm * 32 + mi * 16 + hw * 8 + r;
        float x  = acc[mi][ni][r];
        float xp = __shfl_xor(x, 1, 32);      // RoPE partner (d even<->odd)
        int b = mg >> 11;                     // / T_
        int t = mg & (T_ - 1);
        if (mode == 0 || ng < C_) {
          int h = ng >> 6, d = ng & 63;
          int i2 = d >> 1;
          float w = __expf(-((float)(2 * i2 + 1)) * (LOG_THETA / (float)D_));
          float ang = (float)(t + 1) * w;
          float s, c;
          __sincosf(ang, &s, &c);
          float o = (lane & 1) ? (xp * s + x * c) : (x * c - xp * s);
          outK[(((size_t)(b * H_ + h)) * T_ + t) * D_ + d] = f2bf(o);
        } else {
          int cc = ng - C_;
          int h = cc >> 6, d = cc & 63;
          outV[(((size_t)(b * H_ + h)) * D_ + d) * T_ + t] = f2bf(x);  // v^T
        }
      }
    }
  }
}

// ---- flash attention: 1 wave per (b, h, 32 q-rows) --------------------
// 2 m-tiles share each K-fragment and V-fragment load -> half the L2
// traffic per flop vs a 16-row wave.
__global__ __launch_bounds__(32) void flash_attn(
    const unsigned short* __restrict__ Q, const unsigned short* __restrict__ K,
    const unsigned short* __restrict__ VT,
    const unsigned char* __restrict__ qmask, const unsigned char* __restrict__ kvmask,
    unsigned short* __restrict__ Y)
{
  __shared__ unsigned short Ps[32][40];
  const int lane = threadIdx.x;
  const int l15  = lane & 15;
  const int hw   = lane >> 4;
  const int qt   = blockIdx.x, h = blockIdx.y, b = blockIdx.z;
  const int bh   = b * H_ + h;
  const int t0   = qt * 32;

  Frag qf[2][2];
  #pragma unroll
  for (int mi = 0; mi < 2; ++mi)
    #pragma unroll
    for (int ks = 0; ks < 2; ++ks) {
      const unsigned short* p =
          Q + ((size_t)bh * T_ + t0 + mi * 16 + l15) * D_ + ks * 32 + hw * 8;
      qf[mi][ks].q[0] = *(const v4u*)p;
      qf[mi][ks].q[1] = *(const v4u*)(p + 16);
    }
  float qmr[2][8];
  #pragma unroll
  for (int mi = 0; mi < 2; ++mi)
    #pragma unroll
    for (int r = 0; r < 8; ++r)
      qmr[mi][r] = qmask[(size_t)b * T_ + t0 + mi * 16 + hw * 8 + r] ? 1.f : 0.f;

  v8f o[2][4];
  #pragma unroll
  for (int mi = 0; mi < 2; ++mi)
    #pragma unroll
    for (int ni = 0; ni < 4; ++ni) o[mi][ni] = {};
  float m8[2][8], l8[2][8];
  #pragma unroll
  for (int mi = 0; mi < 2; ++mi)
    #pragma unroll
    for (int r = 0; r < 8; ++r) { m8[mi][r] = -3.0e38f; l8[mi][r] = 0.f; }

  for (int j = 0; j < T_; j += 32) {
    if (j + 32 < T_) {                        // global_prefetch_b8 next chunk
      __builtin_prefetch(K + ((size_t)bh * T_ + j + 32 + lane) * D_, 0, 1);
      __builtin_prefetch(VT + ((size_t)bh * D_ + lane) * T_ + j + 32, 0, 1);
    }
    Frag kf[2][2];
    float kvm[2];
    #pragma unroll
    for (int nc = 0; nc < 2; ++nc) {
      int kr = j + nc * 16 + l15;
      kvm[nc] = kvmask[(size_t)b * T_ + kr] ? 1.f : 0.f;
      #pragma unroll
      for (int ks = 0; ks < 2; ++ks) {
        const unsigned short* p = K + ((size_t)bh * T_ + kr) * D_ + ks * 32 + hw * 8;
        kf[nc][ks].q[0] = *(const v4u*)p;
        kf[nc][ks].q[1] = *(const v4u*)(p + 16);
      }
    }
    #pragma unroll
    for (int mi = 0; mi < 2; ++mi) {
      v8f s[2];
      #pragma unroll
      for (int nc = 0; nc < 2; ++nc) {
        s[nc] = {};
        s[nc] = __builtin_amdgcn_wmma_f32_16x16x32_bf16(
            false, qf[mi][0].v, false, kf[nc][0].v, (short)0, s[nc], false, false);
        s[nc] = __builtin_amdgcn_wmma_f32_16x16x32_bf16(
            false, qf[mi][1].v, false, kf[nc][1].v, (short)0, s[nc], false, false);
      }
      // online softmax; row r lives in vgpr r (rows r / r+8 per half-wave)
      #pragma unroll
      for (int r = 0; r < 8; ++r) {
        float s0 = s[0][r] * 0.125f;                    // 1/sqrt(64)
        float s1 = s[1][r] * 0.125f;
        s0 = (kvm[0] != 0.f) ? s0 : -1.0e9f;
        s1 = (kvm[1] != 0.f) ? s1 : -1.0e9f;
        if (qmr[mi][r] == 0.f) { s0 = 0.f; s1 = 0.f; }  // masked q row -> uniform
        float mx = fmaxf(s0, s1);
        #pragma unroll
        for (int w = 1; w < 16; w <<= 1) mx = fmaxf(mx, __shfl_xor(mx, w, 16));
        float mnew  = fmaxf(m8[mi][r], mx);
        float alpha = __expf(m8[mi][r] - mnew);
        float p0 = __expf(s0 - mnew);
        float p1 = __expf(s1 - mnew);
        float rs = p0 + p1;
        #pragma unroll
        for (int w = 1; w < 16; w <<= 1) rs += __shfl_xor(rs, w, 16);
        l8[mi][r] = l8[mi][r] * alpha + rs;
        m8[mi][r] = mnew;
        #pragma unroll
        for (int ni = 0; ni < 4; ++ni) o[mi][ni][r] *= alpha;
        int row = mi * 16 + hw * 8 + r;
        Ps[row][l15]      = f2bf(p0);                   // transpose P via LDS
        Ps[row][16 + l15] = f2bf(p1);
      }
    }
    asm volatile("s_wait_dscnt 0" ::: "memory");
    Frag vf[4];
    #pragma unroll
    for (int ni = 0; ni < 4; ++ni) {
      const unsigned short* p =
          VT + ((size_t)bh * D_ + ni * 16 + l15) * T_ + j + hw * 8;
      vf[ni].q[0] = *(const v4u*)p;
      vf[ni].q[1] = *(const v4u*)(p + 16);
    }
    #pragma unroll
    for (int mi = 0; mi < 2; ++mi) {
      Frag pf;
      int kb = hw * 8;
      pf.q[0] = *(const v4u*)&Ps[mi * 16 + l15][kb];
      pf.q[1] = *(const v4u*)&Ps[mi * 16 + l15][kb + 16];
      #pragma unroll
      for (int ni = 0; ni < 4; ++ni)
        o[mi][ni] = __builtin_amdgcn_wmma_f32_16x16x32_bf16(
            false, pf.v, false, vf[ni].v, (short)0, o[mi][ni], false, false);
    }
    // per-wave DS ops are in-order: next-iter Ps stores cannot pass these loads
  }
  #pragma unroll
  for (int mi = 0; mi < 2; ++mi)
    #pragma unroll
    for (int ni = 0; ni < 4; ++ni)
      #pragma unroll
      for (int r = 0; r < 8; ++r) {
        float val = o[mi][ni][r] / l8[mi][r];
        int t = t0 + mi * 16 + hw * 8 + r;
        int d = ni * 16 + l15;
        Y[((size_t)b * T_ + t) * C_ + h * D_ + d] = f2bf(val);
      }
}

// ---- output GEMM: out(f32) = Y(bf16, MxK) * W_c(KxN f32) --------------
__global__ __launch_bounds__(256) void gemm_final(
    const unsigned short* __restrict__ Abf, const float* __restrict__ W,
    float* __restrict__ Out)
{
  __shared__ unsigned short As[128][40];
  __shared__ unsigned short Bs[128][40];

  const int tid  = threadIdx.x;
  const int lane = tid & 31;
  const int wave = tid >> 5;
  const int wm   = wave >> 1, wn = wave & 1;
  const int m0   = blockIdx.y * 128;
  const int n0   = blockIdx.x * 128;
  const int hw   = lane >> 4;
  const int l15  = lane & 15;

  v8f acc[2][4];
  #pragma unroll
  for (int a = 0; a < 2; ++a)
    #pragma unroll
    for (int b = 0; b < 4; ++b) acc[a][b] = {};

  v4u ya[2];
  v4f rw[4];
  auto loadregs = [&](int kt) {
    #pragma unroll
    for (int i = 0; i < 2; ++i)
      ya[i] = *(const v4u*)(Abf + (size_t)(m0 + (tid >> 2) + i * 64) * C_ +
                            kt * 32 + (tid & 3) * 8);
    #pragma unroll
    for (int i = 0; i < 4; ++i)
      rw[i] = *(const v4f*)(W + (size_t)(kt * 32 + (tid >> 5) + i * 8) * C_ +
                            n0 + (tid & 31) * 4);
  };
  auto storelds = [&]() {
    #pragma unroll
    for (int i = 0; i < 2; ++i)
      *(v4u*)&As[(tid >> 2) + i * 64][(tid & 3) * 8] = ya[i];
    #pragma unroll
    for (int i = 0; i < 4; ++i) {
      int kr = (tid >> 5) + i * 8, nc = (tid & 31) * 4;
      Bs[nc + 0][kr] = f2bf(rw[i].x); Bs[nc + 1][kr] = f2bf(rw[i].y);
      Bs[nc + 2][kr] = f2bf(rw[i].z); Bs[nc + 3][kr] = f2bf(rw[i].w);
    }
  };

  constexpr int KT = C_ / 32;
  loadregs(0);
  for (int kt = 0; kt < KT; ++kt) {
    storelds();
    __syncthreads();
    if (kt + 1 < KT) loadregs(kt + 1);

    Frag af[2], bfr[4];
    #pragma unroll
    for (int mi = 0; mi < 2; ++mi) {
      int row = wm * 32 + mi * 16 + l15;
      int kb  = hw * 8;
      af[mi].q[0] = *(const v4u*)&As[row][kb];
      af[mi].q[1] = *(const v4u*)&As[row][kb + 16];
    }
    #pragma unroll
    for (int ni = 0; ni < 4; ++ni) {
      int col = wn * 64 + ni * 16 + l15;
      int kb  = hw * 8;
      bfr[ni].q[0] = *(const v4u*)&Bs[col][kb];
      bfr[ni].q[1] = *(const v4u*)&Bs[col][kb + 16];
    }
    #pragma unroll
    for (int mi = 0; mi < 2; ++mi)
      #pragma unroll
      for (int ni = 0; ni < 4; ++ni)
        acc[mi][ni] = __builtin_amdgcn_wmma_f32_16x16x32_bf16(
            false, af[mi].v, false, bfr[ni].v, (short)0, acc[mi][ni], false, false);
    __syncthreads();
  }
  #pragma unroll
  for (int mi = 0; mi < 2; ++mi)
    #pragma unroll
    for (int ni = 0; ni < 4; ++ni) {
      int ng = n0 + wn * 64 + ni * 16 + l15;
      #pragma unroll
      for (int r = 0; r < 8; ++r) {
        int mg = m0 + wm * 32 + mi * 16 + hw * 8 + r;
        Out[(size_t)mg * C_ + ng] = acc[mi][ni][r];
      }
    }
}

// ---- launcher ---------------------------------------------------------
extern "C" void kernel_launch(void* const* d_in, const int* in_sizes, int n_in,
                              void* d_out, int out_size, void* d_ws, size_t ws_size,
                              hipStream_t stream) {
  (void)in_sizes; (void)n_in; (void)out_size; (void)ws_size;
  const float* x_q  = (const float*)d_in[0];
  const float* x_kv = (const float*)d_in[1];
  const unsigned char* qm  = (const unsigned char*)d_in[2];
  const unsigned char* kvm = (const unsigned char*)d_in[3];
  const float* W_q  = (const float*)d_in[4];
  const float* W_kv = (const float*)d_in[5];
  const float* W_c  = (const float*)d_in[6];
  float* out = (float*)d_out;

  const size_t elems = (size_t)B_ * H_ * T_ * D_;       // 8 Mi elements
  unsigned short* qbf = (unsigned short*)d_ws;
  unsigned short* kbf = qbf + elems;
  unsigned short* vT  = kbf + elems;
  unsigned short* ybf = vT  + elems;                     // 64 MiB total ws

  gemm_proj<<<dim3(C_ / 128, (B_ * T_) / 128), 256, 0, stream>>>(
      x_q, W_q, qbf, nullptr, C_, 0);
  gemm_proj<<<dim3((2 * C_) / 128, (B_ * T_) / 128), 256, 0, stream>>>(
      x_kv, W_kv, kbf, vT, 2 * C_, 1);
  flash_attn<<<dim3(T_ / 32, H_, B_), 32, 0, stream>>>(
      qbf, kbf, vT, qm, kvm, ybf);
  gemm_final<<<dim3(C_ / 128, (B_ * T_) / 128), 256, 0, stream>>>(
      ybf, W_c, out);
}
